// ALIKED_2594160246861
// MI455X (gfx1250) — matrix operand order repaired and checked by reading, not compiled, not run on hardware
//
#include <hip/hip_runtime.h>
#include <hip/hip_fp16.h>

// ---------------------------------------------------------------------------
// ALIKED keypoint pipeline for MI455X (gfx1250, wave32).
//   scores_map: (4,1,1536,1536) fp32  ->  out: (4,8192,4) fp32
// k_nms: float4 register-blocked 5x5 NMS (3.75 b128 loads / pixel) with
//        LDS-staged candidate aggregation (1 global atomic per block).
// k_refine: moment sums on v_wmma_f32_16x16x32_f16 (16 keypoints / wave).
// ---------------------------------------------------------------------------

typedef __attribute__((ext_vector_type(16))) _Float16 v16h;
typedef __attribute__((ext_vector_type(8)))  float    v8f;

static constexpr int B_    = 4;
static constexpr int H_    = 1536;
static constexpr int W_    = 1536;
static constexpr int HW_   = H_ * W_;
static constexpr int RAD   = 2;
static constexpr int KTOP  = 8192;
static constexpr int CAP   = 131072;   // per-batch candidate capacity (~94k expected)
static constexpr int NBINS = 4096;
static constexpr float INV_TEMP = 10.0f;  // 1/0.1
static constexpr int BDIM  = 256;         // 8 waves per block on wave32
static constexpr int MAXC  = 256;         // LDS candidate staging per block

// --------------------------- init scratch ----------------------------------
__global__ void k_init(int* __restrict__ hist, int* __restrict__ misc,
                       float* __restrict__ sel_val, int* __restrict__ sel_idx) {
    int i = blockIdx.x * blockDim.x + threadIdx.x;
    if (i < B_ * NBINS) hist[i] = 0;
    if (i < 6 * B_)     misc[i] = 0;
    if (i < B_ * KTOP) { sel_val[i] = 0.0f; sel_idx[i] = 0; }
}

// --------------------------- 5x5 NMS + collect -----------------------------
// 4 pixels per thread; 3 aligned float4 loads per row cover the 12-wide
// column window [x4-4, x4+7]. One global atomic per block via LDS staging.
__global__ void __launch_bounds__(BDIM)
k_nms(const float* __restrict__ s,
      float* __restrict__ cand_val, int* __restrict__ cand_idx,
      int* __restrict__ cand_cnt) {
    __shared__ int   l_cnt;
    __shared__ int   l_base;
    __shared__ int   l_n;
    __shared__ float l_val[MAXC];
    __shared__ int   l_idx[MAXC];

    const int wq  = W_ / 4;                          // 384 quads per row
    const int i   = blockIdx.x * blockDim.x + threadIdx.x;
    const int b   = i / (wq * H_);                   // uniform per block
    const int rem = i - b * (wq * H_);
    const int y   = rem / wq;
    const int x4  = (rem - y * wq) * 4;
    const bool rowvalid = (y >= RAD) && (y < H_ - RAD);

    if (threadIdx.x == 0) l_cnt = 0;
    __syncthreads();

    if (rowvalid) {
        const float* sb  = s + (size_t)b * HW_;
        const float4* rb = (const float4*)(sb + (size_t)(y - RAD) * W_);
        const int q      = x4 >> 2;                  // quad index in row
        const bool hasL  = (x4 >= 4);
        const bool hasR  = (x4 <= W_ - 8);
        const int qL     = hasL ? q - 1 : q;
        const int qR     = hasR ? q + 1 : q;
        const int rowq   = W_ / 4;

        float wmax[4] = {-1e30f, -1e30f, -1e30f, -1e30f};
        float cval[4];

        __builtin_prefetch(sb + (size_t)(y + 3) * W_ + x4, 0, 1); // global_prefetch_b8

        #pragma unroll
        for (int r = 0; r < 5; ++r) {
            float4 cL = rb[(size_t)r * rowq + qL];
            float4 cC = rb[(size_t)r * rowq + q];
            float4 cR = rb[(size_t)r * rowq + qR];
            float w[12] = { cL.x, cL.y, cL.z, cL.w,
                            cC.x, cC.y, cC.z, cC.w,
                            cR.x, cR.y, cR.z, cR.w };
            #pragma unroll
            for (int j = 0; j < 4; ++j) {
                float m = w[j + 2];
                m = fmaxf(m, w[j + 3]);
                m = fmaxf(m, w[j + 4]);
                m = fmaxf(m, w[j + 5]);
                m = fmaxf(m, w[j + 6]);
                wmax[j] = fmaxf(wmax[j], m);
                if (r == RAD) cval[j] = w[j + 4];
            }
        }

        #pragma unroll
        for (int j = 0; j < 4; ++j) {
            int x = x4 + j;
            bool cand = (x >= RAD) && (x < W_ - RAD) && (cval[j] >= wmax[j]);
            if (cand) {
                int p = y * W_ + x;
                int t = atomicAdd(&l_cnt, 1);
                if (t < MAXC) {
                    l_val[t] = cval[j];
                    l_idx[t] = p;
                } else {                               // rare spill path
                    int pos = atomicAdd(&cand_cnt[b], 1);
                    if (pos < CAP) {
                        cand_val[b * CAP + pos] = cval[j];
                        cand_idx[b * CAP + pos] = p;
                    }
                }
            }
        }
    }
    __syncthreads();
    if (threadIdx.x == 0) {
        int n = min(l_cnt, MAXC);
        l_n = n;
        l_base = n ? atomicAdd(&cand_cnt[b], n) : 0;
    }
    __syncthreads();
    for (int t = threadIdx.x; t < l_n; t += blockDim.x) {
        int pos = l_base + t;
        if (pos < CAP) {
            cand_val[b * CAP + pos] = l_val[t];
            cand_idx[b * CAP + pos] = l_idx[t];
        }
    }
}

// --------------------------- histogram of candidates -----------------------
__global__ void k_hist(const float* __restrict__ cand_val,
                       const int* __restrict__ cand_cnt, int* __restrict__ hist) {
    int b = blockIdx.y;
    int n = min(cand_cnt[b], CAP);
    for (int i = blockIdx.x * blockDim.x + threadIdx.x; i < n;
         i += gridDim.x * blockDim.x) {
        float v = cand_val[b * CAP + i];
        int bin = (int)(v * (float)NBINS);
        bin = max(0, min(NBINS - 1, bin));
        atomicAdd(&hist[b * NBINS + bin], 1);
    }
}

// --------------------------- threshold bin selection -----------------------
__global__ void k_select(const int* __restrict__ hist, int* __restrict__ thr_bin,
                         int* __restrict__ num_above, int* __restrict__ take_eq) {
    int b = blockIdx.x;
    if (threadIdx.x != 0) return;
    int cum = 0, tb = -1, na = 0, tk = 0;
    for (int bin = NBINS - 1; bin >= 0; --bin) {
        int c = hist[b * NBINS + bin];
        if (cum + c >= KTOP) { tb = bin; na = cum; tk = KTOP - cum; break; }
        cum += c;
    }
    thr_bin[b] = tb; num_above[b] = na; take_eq[b] = tk;
}

// --------------------------- compact top-K ---------------------------------
// Wave-aggregated counter atomics (uniform trip-count keeps EXEC full).
__global__ void k_compact(const float* __restrict__ cand_val,
                          const int* __restrict__ cand_idx,
                          const int* __restrict__ cand_cnt,
                          const int* __restrict__ thr_bin,
                          const int* __restrict__ num_above,
                          const int* __restrict__ take_eq,
                          int* __restrict__ cnt_hi, int* __restrict__ cnt_eq,
                          float* __restrict__ sel_val, int* __restrict__ sel_idx) {
    int b  = blockIdx.y;
    int n  = min(cand_cnt[b], CAP);
    int tb = thr_bin[b];
    int na = num_above[b];
    int tk = take_eq[b];
    int lane   = threadIdx.x & 31;
    int stride = gridDim.x * blockDim.x;
    for (int i = blockIdx.x * blockDim.x + threadIdx.x; (i - lane) < n; i += stride) {
        bool valid = i < n;
        float v = 0.0f;
        int bin = -2;
        if (valid) {
            v   = cand_val[b * CAP + i];
            bin = max(0, min(NBINS - 1, (int)(v * (float)NBINS)));
        }
        bool hic = valid && (bin > tb);
        bool eqc = valid && (bin == tb);

        unsigned long long mh = __ballot(hic);
        unsigned long long me = __ballot(eqc);
        int ch = __popcll(mh);
        int ce = __popcll(me);
        int bh = 0, be = 0;
        if (lane == 0) {
            if (ch) bh = atomicAdd(&cnt_hi[b], ch);
            if (ce) be = atomicAdd(&cnt_eq[b], ce);
        }
        bh = __shfl(bh, 0);
        be = __shfl(be, 0);
        unsigned long long below = (1ull << lane) - 1ull;

        int slot = -1;
        if (hic) {
            slot = bh + __popcll(mh & below);
        } else if (eqc) {
            int t = be + __popcll(me & below);
            if (t < tk) slot = na + t;
        }
        if (slot >= 0 && slot < KTOP) {
            sel_val[b * KTOP + slot] = v;
            sel_idx[b * KTOP + slot] = cand_idx[b * CAP + i];
        }
    }
}

// --------------------------- WMMA refinement -------------------------------
// Per wave: 16 keypoints. A (16x32 f16) = patch exponentials (25 used, pad 0).
// B (32x16 f16) = moment matrix: col0=1, col1=dx, col2=dy, col3=dx^2+dy^2.
// D (16x16 f32) -> S0=Σe, S1=Σe·dx, S2=Σe·dy, S3=Σe·r².
__global__ void __launch_bounds__(BDIM)
k_refine(const float* __restrict__ s, const float* __restrict__ sel_val,
         const int* __restrict__ sel_idx, float* __restrict__ out) {
    __shared__ float lds[BDIM / 32][16][17];
    const int wave = threadIdx.x >> 5;
    const int lane = threadIdx.x & 31;
    const int mloc = lane & 15;
    const bool hi  = lane >= 16;
    const int base = blockIdx.x * (BDIM / 32) * 16 + wave * 16; // flat b*KTOP+k
    const int kg   = base + mloc;
    const int b    = kg / KTOP;
    const int idx  = sel_idx[kg];
    const float maxv = sel_val[kg];       // NMS local max == patch max
    const int ky = idx / W_;
    const int kx = idx - ky * W_;
    const float* sb = s + (size_t)b * HW_;

    // A operand: lane<16 holds K=0..7,16..23 of row mloc; lane>=16 K=8..15,24..31
    v16h a;
    #pragma unroll
    for (int h = 0; h < 16; ++h) {
        int p = hi ? (h < 8 ? h + 8 : h + 16) : (h < 8 ? h : h + 8);
        float e = 0.0f;
        if (p < 25) {
            int dy = p / 5 - RAD;
            int dx = p - (p / 5) * 5 - RAD;
            int yy = min(max(ky + dy, 0), H_ - 1);
            int xx = min(max(kx + dx, 0), W_ - 1);
            e = __expf((sb[yy * W_ + xx] - maxv) * INV_TEMP);
        }
        a[h] = (_Float16)e;
    }

    // B operand: lanes 0-15 hold K=0..7,16..23, lanes 16-31 hold K=8..15,24..31; N = lane&15
    v16h bm;
    #pragma unroll
    for (int h = 0; h < 16; ++h) {
        int kk = hi ? (h < 8 ? h + 8 : h + 16) : (h < 8 ? h : h + 8);
        float w = 0.0f;
        if (kk < 25 && mloc < 4) {
            int dy = kk / 5 - RAD;
            int dx = kk - (kk / 5) * 5 - RAD;
            w = (mloc == 0) ? 1.0f
              : (mloc == 1) ? (float)dx
              : (mloc == 2) ? (float)dy
                            : (float)(dx * dx + dy * dy);
        }
        bm[h] = (_Float16)w;
    }

    v8f c = {};
    c = __builtin_amdgcn_wmma_f32_16x16x32_f16(false, a, false, bm,
                                               (short)0, c, false, false);

    // D layout: lane L, VGPR r -> D[r + (L<16?0:8)][L&15]
    #pragma unroll
    for (int r = 0; r < 8; ++r)
        lds[wave][r + (hi ? 8 : 0)][mloc] = c[r];
    __syncthreads();

    if (lane < 16) {
        float S0 = lds[wave][lane][0];
        float S1 = lds[wave][lane][1];
        float S2 = lds[wave][lane][2];
        float S3 = lds[wave][lane][3];
        int kg2  = base + lane;
        int b2   = kg2 / KTOP;
        int idx2 = sel_idx[kg2];
        int ky2  = idx2 / W_;
        int kx2  = idx2 - ky2 * W_;
        float inv = 1.0f / S0;
        float xr  = S1 * inv;
        float yr  = S2 * inv;
        float disp = (S3 * inv - xr * xr - yr * yr) * (1.0f / (RAD * RAD));
        float px = (float)kx2 + xr;
        float py = (float)ky2 + yr;
        float kpnx = px * (2.0f / (float)(W_ - 1)) - 1.0f;
        float kpny = py * (2.0f / (float)(H_ - 1)) - 1.0f;
        int x0 = min(max((int)floorf(px), 0), W_ - 2);
        int y0 = min(max((int)floorf(py), 0), H_ - 2);
        float wx = px - (float)x0;
        float wy = py - (float)y0;
        const float* sb2 = s + (size_t)b2 * HW_;
        float v00 = sb2[y0 * W_ + x0];
        float v01 = sb2[y0 * W_ + x0 + 1];
        float v10 = sb2[(y0 + 1) * W_ + x0];
        float v11 = sb2[(y0 + 1) * W_ + x0 + 1];
        float sc = (1.0f - wx) * (1.0f - wy) * v00 + wx * (1.0f - wy) * v01 +
                   (1.0f - wx) * wy * v10 + wx * wy * v11;
        float* o = out + (size_t)kg2 * 4;
        o[0] = kpnx; o[1] = kpny; o[2] = sc; o[3] = disp;
    }
}

// --------------------------- launch ----------------------------------------
extern "C" void kernel_launch(void* const* d_in, const int* in_sizes, int n_in,
                              void* d_out, int out_size, void* d_ws, size_t ws_size,
                              hipStream_t stream) {
    (void)in_sizes; (void)n_in; (void)out_size; (void)ws_size;
    const float* s = (const float*)d_in[0];
    float* out = (float*)d_out;

    char* ws = (char*)d_ws;
    size_t o = 0;
    float* cand_val = (float*)(ws + o); o += (size_t)B_ * CAP * 4;
    int*   cand_idx = (int*)  (ws + o); o += (size_t)B_ * CAP * 4;
    float* sel_val  = (float*)(ws + o); o += (size_t)B_ * KTOP * 4;
    int*   sel_idx  = (int*)  (ws + o); o += (size_t)B_ * KTOP * 4;
    int*   hist     = (int*)  (ws + o); o += (size_t)B_ * NBINS * 4;
    int*   misc     = (int*)  (ws + o); o += (size_t)6 * B_ * 4;
    int* cand_cnt  = misc;
    int* cnt_hi    = misc + B_;
    int* cnt_eq    = misc + 2 * B_;
    int* thr_bin   = misc + 3 * B_;
    int* num_above = misc + 4 * B_;
    int* take_eq   = misc + 5 * B_;

    k_init<<<(B_ * KTOP + BDIM - 1) / BDIM, BDIM, 0, stream>>>(hist, misc, sel_val, sel_idx);
    k_nms<<<(B_ * H_ * (W_ / 4)) / BDIM, BDIM, 0, stream>>>(s, cand_val, cand_idx, cand_cnt);
    k_hist<<<dim3(64, B_), BDIM, 0, stream>>>(cand_val, cand_cnt, hist);
    k_select<<<B_, 32, 0, stream>>>(hist, thr_bin, num_above, take_eq);
    k_compact<<<dim3(64, B_), BDIM, 0, stream>>>(cand_val, cand_idx, cand_cnt,
                                                 thr_bin, num_above, take_eq,
                                                 cnt_hi, cnt_eq, sel_val, sel_idx);
    k_refine<<<(B_ * KTOP) / ((BDIM / 32) * 16), BDIM, 0, stream>>>(s, sel_val, sel_idx, out);
}
